// GraphormerMultiheadAttention_29489245454851
// MI455X (gfx1250) — compile-verified, matrix operand order checked by reading, not run on hardware
//
#include <hip/hip_runtime.h>
#include <hip/hip_bf16.h>

// ---------------------------------------------------------------------------
// Types for CDNA5 WMMA (wave32): v_wmma_f32_16x16x32_bf16
// ---------------------------------------------------------------------------
typedef __bf16 bf16_t;
typedef __attribute__((ext_vector_type(16))) __bf16 v16bf;
typedef __attribute__((ext_vector_type(8)))  __bf16 bf16x8;
typedef __attribute__((ext_vector_type(8)))  float  v8f;
typedef __attribute__((ext_vector_type(4)))  unsigned u32x4;
typedef __attribute__((ext_vector_type(8)))  unsigned u32x8;

union FragU { v16bf v; bf16x8 h[2]; };

__device__ __forceinline__ v8f wmma_bf16(v16bf a, v16bf b, v8f c) {
    return __builtin_amdgcn_wmma_f32_16x16x32_bf16(
        false, a, false, b, (short)0, c, false, false);
}

__device__ __forceinline__ bf16_t f2bf(float f) {
    union { float f; unsigned u; } x; x.f = f;
    unsigned r = x.u + 0x7FFFu + ((x.u >> 16) & 1u);   // round-to-nearest-even
    unsigned short hs = (unsigned short)(r >> 16);
    union { unsigned short s; bf16_t b; } y; y.s = hs;
    return y.b;
}

// Fragment load: lane holds row (lane&15), two 16B K-chunks at hi8 / hi8+16.
__device__ __forceinline__ v16bf frag_ld(const bf16_t* p, int hi8) {
    FragU u;
    u.h[0] = *(const bf16x8*)(p + hi8);
    u.h[1] = *(const bf16x8*)(p + hi8 + 16);
    return u.v;
}

// Low 32 bits of a generic pointer to __shared__ = workgroup LDS byte offset.
__device__ __forceinline__ unsigned lds_lo32(const void* p) {
    return (unsigned)(uintptr_t)p;
}

// ---------------------------------------------------------------------------
// Tensor Data Mover: issue a 2D tile DMA (global bf16 -> LDS).
// Tile = 128 rows x 32 cols of 2-byte elements, row stride = K elements.
// D# group0: count=1 | lds_addr | global_addr(57b) | type=2.
// D# group1: data_size=1(2B), tensor_dim0=K, tensor_dim1=128,
//            tile_dim0=32, tile_dim1=128, tensor_dim0_stride=K.
// Issued from one wave; completion via s_wait_tensorcnt (in-order).
// ---------------------------------------------------------------------------
__device__ __forceinline__ void tdm_load_tile_2d(const bf16_t* gsrc,
                                                 unsigned lds_byte_addr,
                                                 unsigned K)
{
    unsigned long long ga = (unsigned long long)(uintptr_t)gsrc;
    u32x4 g0;
    g0[0] = 1u;                                   // count=1, user mode
    g0[1] = lds_byte_addr;                        // lds_addr [63:32]
    g0[2] = (unsigned)ga;                         // global_addr [95:64]
    g0[3] = (unsigned)(ga >> 32) | (2u << 30);    // global_addr hi | type=2

    const unsigned td0 = K, td1 = 128u;
    const unsigned tile0 = 32u, tile1 = 128u, tile2 = 0u;
    const unsigned long long st0 = (unsigned long long)K;

    u32x8 g1;
    g1[0] = 0x00010000u;                          // wg_mask=0, data_size=1 (2B)
    g1[1] = (td0 & 0xFFFFu) << 16;                // tensor_dim0[15:0] @ [63:48]
    g1[2] = (td0 >> 16) | ((td1 & 0xFFFFu) << 16);// td0 hi | tensor_dim1 lo
    g1[3] = (td1 >> 16) | (tile0 << 16);          // td1 hi | tile_dim0
    g1[4] = tile1 | (tile2 << 16);                // tile_dim1 | tile_dim2=0
    g1[5] = (unsigned)(st0 & 0xFFFFFFFFu);        // stride0 [31:0]
    g1[6] = (unsigned)(st0 >> 32);                // stride0 [47:32] | stride1 lo=0
    g1[7] = 0u;                                   // stride1 hi = 0

    asm volatile("tensor_load_to_lds %0, %1" :: "s"(g0), "s"(g1) : "memory");
}

// ---------------------------------------------------------------------------
// Streaming fp32 -> bf16 conversion (one pass; removes all conversion work
// from the GEMM hot loops and dedups it across tiles).
// ---------------------------------------------------------------------------
__global__ __launch_bounds__(256) void cvt_f32_bf16(const float* __restrict__ src,
                                                    bf16_t* __restrict__ dst,
                                                    int n)
{
    int i = ((int)blockIdx.x * 256 + (int)threadIdx.x) * 8;
    if (i >= n) return;
    float4 a = *(const float4*)(src + i);
    float4 b = *(const float4*)(src + i + 4);
    bf16x8 o;
    o[0]=f2bf(a.x); o[1]=f2bf(a.y); o[2]=f2bf(a.z); o[3]=f2bf(a.w);
    o[4]=f2bf(b.x); o[5]=f2bf(b.y); o[6]=f2bf(b.z); o[7]=f2bf(b.w);
    *(bf16x8*)(dst + i) = o;
}

// ---------------------------------------------------------------------------
// GEMM: C[M,N] = A[M,K](bf16) @ W[N,K](bf16)^T + bias[N](fp32)
//   OUT_MODE 0: bf16 row-major [M,N]
//   OUT_MODE 1: bf16 V^T layout: [(m/1024), n, (m%1024)]  (for P@V stage)
//   OUT_MODE 2: fp32 row-major [M,N] (final output, nontemporal)
// 256 threads / 8 waves, tile 128x128, K-step 32, TDM double-buffered LDS.
// Wave w -> 2x4 grid of 16x16 WMMA tiles.
// ---------------------------------------------------------------------------
template<int OUT_MODE>
__global__ __launch_bounds__(256) void gemm_bias_wmma(
    const bf16_t* __restrict__ A, const bf16_t* __restrict__ W,
    const float* __restrict__ bias, void* __restrict__ Cptr,
    int M, int N, int K)
{
    constexpr int SCOL = 1024;
    __shared__ __align__(16) bf16_t lA[2][128 * 32];
    __shared__ __align__(16) bf16_t lB[2][128 * 32];

    const int tid   = threadIdx.x;
    const int nT    = N / 128;
    const int mBase = (int)(blockIdx.x / nT) * 128;
    const int nBase = (int)(blockIdx.x % nT) * 128;

    const int wave = tid >> 5;
    const int lane = tid & 31;
    const int wr   = wave >> 1;         // 0..3 -> +wr*32 rows
    const int wc   = wave & 1;          // 0..1 -> +wc*64 cols
    const int rlo  = lane & 15;
    const int hi8  = (lane >> 4) << 3;

    const bf16_t* Atile = A + (size_t)mBase * K;
    const bf16_t* Wtile = W + (size_t)nBase * K;

    v8f acc[2][4] = {};

    // Prologue: DMA k-tile 0 into buffer 0 (wave 0 issues; TDM ignores EXEC).
    if (wave == 0) {
        tdm_load_tile_2d(Atile, lds_lo32(&lA[0][0]), (unsigned)K);
        tdm_load_tile_2d(Wtile, lds_lo32(&lB[0][0]), (unsigned)K);
    }

    int buf = 0;
    for (int k0 = 0; k0 < K; k0 += 32) {
        if (wave == 0) {
            if (k0 + 32 < K) {
                // Prefetch next k-tile into the other buffer, then retire the
                // current pair (TDM completes in order: <=2 outstanding).
                tdm_load_tile_2d(Atile + k0 + 32, lds_lo32(&lA[buf ^ 1][0]),
                                 (unsigned)K);
                tdm_load_tile_2d(Wtile + k0 + 32, lds_lo32(&lB[buf ^ 1][0]),
                                 (unsigned)K);
                asm volatile("s_wait_tensorcnt 0x2" ::: "memory");
            } else {
                asm volatile("s_wait_tensorcnt 0x0" ::: "memory");
            }
        }
        __syncthreads();    // current buffer now valid for all waves

        v16bf af[2], bfr[4];
        #pragma unroll
        for (int i = 0; i < 2; ++i)
            af[i] = frag_ld(&lA[buf][(wr * 32 + i * 16 + rlo) * 32], hi8);
        #pragma unroll
        for (int j = 0; j < 4; ++j)
            bfr[j] = frag_ld(&lB[buf][(wc * 64 + j * 16 + rlo) * 32], hi8);

        #pragma unroll
        for (int i = 0; i < 2; ++i)
            #pragma unroll
            for (int j = 0; j < 4; ++j)
                acc[i][j] = wmma_bf16(af[i], bfr[j], acc[i][j]);

        __syncthreads();    // all waves done reading buf before next overwrite
        buf ^= 1;
    }

    // ---- epilogue: bias add + store ----
    #pragma unroll
    for (int i = 0; i < 2; ++i) {
        #pragma unroll
        for (int j = 0; j < 4; ++j) {
            const int ncol = nBase + wc * 64 + j * 16 + rlo;
            const float bv = bias[ncol];
            #pragma unroll
            for (int vi = 0; vi < 8; ++vi) {
                const int mrow = mBase + wr * 32 + i * 16 + vi + hi8;
                const float val = acc[i][j][vi] + bv;
                if (OUT_MODE == 0) {
                    ((bf16_t*)Cptr)[(size_t)mrow * N + ncol] = f2bf(val);
                } else if (OUT_MODE == 1) {
                    const int bb = mrow / SCOL;
                    const int ss = mrow % SCOL;
                    ((bf16_t*)Cptr)[((size_t)bb * N + ncol) * SCOL + ss] = f2bf(val);
                } else {
                    __builtin_nontemporal_store(
                        val, &((float*)Cptr)[(size_t)mrow * N + ncol]);
                }
            }
        }
    }
}

// ---------------------------------------------------------------------------
// Flash attention (bf16 WMMA, fp32 online softmax).
// Block = 8 waves; block owns one (b, h, 128-row t-tile); wave owns 16 t-rows.
// K/V fragments stream straight from global into WMMA (layouts line up);
// P re-swizzles through a per-wave LDS region (s_wait_dscnt for the RAW).
// edge_bias / attn_mask are streamed once -> nontemporal loads (save L2 for
// the reused K/V/weight working set; 192MB L2 holds all of it).
// ---------------------------------------------------------------------------
__global__ __launch_bounds__(256) void flash_attn_wmma(
    const bf16_t* __restrict__ q,   // [B, T, E] bf16
    const bf16_t* __restrict__ k,   // [B, S, E] bf16
    const bf16_t* __restrict__ vT,  // [B, E, S] bf16 (pre-transposed)
    const float*  __restrict__ edge,// [B, T, S]
    const float*  __restrict__ amask,// [T, S]
    const unsigned char* __restrict__ kpm, // [B, S]
    bf16_t* __restrict__ out,       // [B, T, E] bf16
    int B, int H, int T, int S, float scale)
{
    const int E = H * 64;
    const int tTiles = T / 128;
    const int bid = blockIdx.x;
    const int tT  = bid % tTiles;
    const int h   = (bid / tTiles) % H;
    const int b   = bid / (tTiles * H);

    const int wave = threadIdx.x >> 5;
    const int lane = threadIdx.x & 31;
    const int rlo  = lane & 15;
    const int hi8  = (lane >> 4) << 3;
    const int tbase = tT * 128 + wave * 16;

    __shared__ __align__(16) bf16_t lp[8 * 16 * 128];
    bf16_t* myP = lp + wave * 16 * 128;

    v16bf qf[2];
    {
        const bf16_t* qp = q + ((size_t)(b * T + tbase + rlo)) * E + h * 64;
        qf[0] = frag_ld(qp, hi8);
        qf[1] = frag_ld(qp + 32, hi8);
    }

    v8f o[4] = {};
    float mrow[8], lrow[8];
    #pragma unroll
    for (int vi = 0; vi < 8; ++vi) { mrow[vi] = -3.0e38f; lrow[vi] = 0.0f; }

    for (int s0 = 0; s0 < S; s0 += 128) {
        // ---- scores: 8 subtiles of 16x16, K-dim = 64 (2 WMMAs each) ----
        v8f sc[8];
        #pragma unroll
        for (int j = 0; j < 8; ++j) {
            const bf16_t* kp = k + ((size_t)(b * S + s0 + j * 16 + rlo)) * E + h * 64;
            v16bf kf0 = frag_ld(kp, hi8);
            v16bf kf1 = frag_ld(kp + 32, hi8);
            v8f t = {};
            t = wmma_bf16(qf[0], kf0, t);
            t = wmma_bf16(qf[1], kf1, t);
            sc[j] = t;
        }
        // ---- scale + attn_mask + edge_bias + key_padding_mask ----
        #pragma unroll
        for (int j = 0; j < 8; ++j) {
            const int scol = s0 + j * 16 + rlo;
            const bool dead = kpm[(size_t)b * S + scol] != 0;
            #pragma unroll
            for (int vi = 0; vi < 8; ++vi) {
                const int trow = tbase + vi + hi8;
                float v = sc[j][vi] * scale
                        + __builtin_nontemporal_load(
                              &amask[(size_t)trow * S + scol])
                        + __builtin_nontemporal_load(
                              &edge[((size_t)(b * T + trow)) * S + scol]);
                sc[j][vi] = dead ? -3.0e38f : v;
            }
        }
        // ---- online softmax per owned row (16-lane half-wave shfl) ----
        #pragma unroll
        for (int vi = 0; vi < 8; ++vi) {
            float mx = sc[0][vi];
            #pragma unroll
            for (int j = 1; j < 8; ++j) mx = fmaxf(mx, sc[j][vi]);
            #pragma unroll
            for (int d = 1; d < 16; d <<= 1)
                mx = fmaxf(mx, __shfl_xor(mx, d, 32));
            const float mnew = fmaxf(mrow[vi], mx);
            const float corr = __expf(mrow[vi] - mnew);
            mrow[vi] = mnew;
            float sum = 0.0f;
            #pragma unroll
            for (int j = 0; j < 8; ++j) {
                const float p = __expf(sc[j][vi] - mnew);
                sc[j][vi] = p;
                sum += p;
            }
            #pragma unroll
            for (int d = 1; d < 16; d <<= 1)
                sum += __shfl_xor(sum, d, 32);
            lrow[vi] = lrow[vi] * corr + sum;
            #pragma unroll
            for (int dd = 0; dd < 4; ++dd) o[dd][vi] *= corr;
        }
        // ---- P -> LDS (bf16) to re-swizzle into A-fragment layout ----
        #pragma unroll
        for (int j = 0; j < 8; ++j)
            #pragma unroll
            for (int vi = 0; vi < 8; ++vi)
                myP[(vi + hi8) * 128 + j * 16 + rlo] = f2bf(sc[j][vi]);
        asm volatile("s_wait_dscnt 0" ::: "memory");   // wave-local LDS RAW
        // ---- O += P(16x128) @ V^T : 4 k-chunks x 4 d-tiles = 16 WMMAs ----
        #pragma unroll
        for (int jk = 0; jk < 4; ++jk) {
            v16bf pf = frag_ld(myP + rlo * 128 + jk * 32, hi8);
            #pragma unroll
            for (int dd = 0; dd < 4; ++dd) {
                const bf16_t* vp = vT + ((size_t)b * E + h * 64 + dd * 16 + rlo) * S
                                   + s0 + jk * 32;
                v16bf vf = frag_ld(vp, hi8);
                o[dd] = wmma_bf16(pf, vf, o[dd]);
            }
        }
    }

    // ---- normalize + store bf16 heads output [B, T, E] ----
    #pragma unroll
    for (int dd = 0; dd < 4; ++dd) {
        #pragma unroll
        for (int vi = 0; vi < 8; ++vi) {
            const int trow = tbase + vi + hi8;
            const float val = o[dd][vi] / lrow[vi];
            out[((size_t)(b * T + trow)) * E + h * 64 + dd * 16 + rlo] = f2bf(val);
        }
    }
}

// ---------------------------------------------------------------------------
// Launch: fp32->bf16 conversions -> Q/K/V projections -> flash attention ->
// output projection.  Workspace (bf16):
//   q | k | vT | attn_out (4 x 8MB)  +  xq | xk | xv (3 x 8MB)
//   + Wq/Wk/Wv/Wo bf16 (4 x 2MB)   = 64 MB.
// ---------------------------------------------------------------------------
extern "C" void kernel_launch(void* const* d_in, const int* in_sizes, int n_in,
                              void* d_out, int out_size, void* d_ws, size_t ws_size,
                              hipStream_t stream) {
    constexpr int Bc = 4, Tc = 1024, Sc = 1024, Ec = 1024, Hc = 16;

    const float* query = (const float*)d_in[0];
    const float* key_  = (const float*)d_in[1];
    const float* value = (const float*)d_in[2];
    const float* edge  = (const float*)d_in[3];
    const float* amask = (const float*)d_in[4];
    const unsigned char* kpm = (const unsigned char*)d_in[5];
    const float* Wq = (const float*)d_in[6];
    const float* bq = (const float*)d_in[7];
    const float* Wk = (const float*)d_in[8];
    const float* bk = (const float*)d_in[9];
    const float* Wv = (const float*)d_in[10];
    const float* bv = (const float*)d_in[11];
    const float* Wo = (const float*)d_in[12];
    const float* bo = (const float*)d_in[13];

    const size_t BTE = (size_t)Bc * Tc * Ec;   // 4M elems
    const size_t EE  = (size_t)Ec * Ec;        // 1M elems
    bf16_t* qb  = (bf16_t*)d_ws;
    bf16_t* kb  = qb  + BTE;
    bf16_t* vTb = kb  + BTE;
    bf16_t* aob = vTb + BTE;
    bf16_t* xq  = aob + BTE;
    bf16_t* xk  = xq  + BTE;
    bf16_t* xv  = xk  + BTE;
    bf16_t* wqb = xv  + BTE;
    bf16_t* wkb = wqb + EE;
    bf16_t* wvb = wkb + EE;
    bf16_t* wob = wvb + EE;

    dim3 blk(256);

    // one-pass fp32 -> bf16 conversions (streaming, bandwidth-bound)
    dim3 cg1((unsigned)(BTE / 8 / 256));
    dim3 cg2((unsigned)(EE / 8 / 256));
    cvt_f32_bf16<<<cg1, blk, 0, stream>>>(query, xq, (int)BTE);
    cvt_f32_bf16<<<cg1, blk, 0, stream>>>(key_,  xk, (int)BTE);
    cvt_f32_bf16<<<cg1, blk, 0, stream>>>(value, xv, (int)BTE);
    cvt_f32_bf16<<<cg2, blk, 0, stream>>>(Wq, wqb, (int)EE);
    cvt_f32_bf16<<<cg2, blk, 0, stream>>>(Wk, wkb, (int)EE);
    cvt_f32_bf16<<<cg2, blk, 0, stream>>>(Wv, wvb, (int)EE);
    cvt_f32_bf16<<<cg2, blk, 0, stream>>>(Wo, wob, (int)EE);

    dim3 gg((Bc * Tc / 128) * (Ec / 128));     // 256 blocks

    gemm_bias_wmma<0><<<gg, blk, 0, stream>>>(xq, wqb, bq, qb,  Bc * Tc, Ec, Ec);
    gemm_bias_wmma<0><<<gg, blk, 0, stream>>>(xk, wkb, bk, kb,  Bc * Sc, Ec, Ec);
    gemm_bias_wmma<1><<<gg, blk, 0, stream>>>(xv, wvb, bv, vTb, Bc * Sc, Ec, Ec);

    dim3 fg(Bc * Hc * (Tc / 128));             // 512 blocks
    flash_attn_wmma<<<fg, blk, 0, stream>>>(qb, kb, vTb, edge, amask, kpm,
                                            aob, Bc, Hc, Tc, Sc, 0.125f);

    gemm_bias_wmma<2><<<gg, blk, 0, stream>>>(aob, wob, bo, d_out, Bc * Tc, Ec, Ec);
}
